// EquidistantDiscreteContinuousConv2d_31155692765838
// MI455X (gfx1250) — matrix operand order repaired and checked by reading, not compiled, not run on hardware
//
#include <hip/hip_runtime.h>

// ---------------- static problem config (matches reference) ----------------
#define H_IMG 1024
#define W_IMG 1024
#define CIN   32
#define COUT  32
#define KS    7       // folded conv kernel is 7x7
#define TAPS  49
#define NK    9       // psi basis functions
#define TILE_H 16     // spatial tile rows per workgroup
#define TILE_W 32     // spatial tile cols per workgroup
#define LROWS (TILE_H + 6)   // 22
#define LCOLS (TILE_W + 6)   // 38
#define CHP   40      // LDS per-pixel pitch in halfs (32 ch + 8 pad) -> 80B, 16B aligned

typedef __attribute__((ext_vector_type(16))) __bf16          v16bf;
typedef __attribute__((ext_vector_type(8)))  float           v8f;

static __device__ __forceinline__ unsigned short f2bf(float f) {
  // round-to-nearest-even fp32 -> bf16
  unsigned int u = __float_as_uint(f);
  u += 0x7FFFu + ((u >> 16) & 1u);
  return (unsigned short)(u >> 16);
}

// ---------------------------------------------------------------------------
// Stage 1: fold psi into weight, emit per-tap WMMA A-tiles in exact lane order.
// aws[((tap*2 + mtile)*32 + lane)*16 + e] = bf16( sum_k psi[k,u,v] * w[o,i,k] )
//   o = mtile*16 + (lane&15)
//   i = (e&7) + (e>>3)*16 + (lane>>4)*8     (A-matrix 16x32 bf16 lane layout)
// ---------------------------------------------------------------------------
__global__ void disco_build_atiles(const float* __restrict__ weight,
                                   const float* __restrict__ psi,
                                   unsigned short* __restrict__ aws) {
  int j = blockIdx.x * blockDim.x + threadIdx.x;
  if (j >= TAPS * 2 * 32) return;
  const int tap   = j >> 6;           // /64
  const int rest  = j & 63;
  const int mtile = rest >> 5;
  const int lane  = rest & 31;
  const int o     = mtile * 16 + (lane & 15);
  const int u     = tap / KS;
  const int v     = tap - u * KS;

  float pk[NK];
#pragma unroll
  for (int k = 0; k < NK; ++k) pk[k] = psi[(k * KS + u) * KS + v];

  unsigned short* dst = aws + (size_t)j * 16;
#pragma unroll
  for (int e = 0; e < 16; ++e) {
    const int i = (e & 7) + ((e >> 3) << 4) + ((lane >> 4) << 3);
    const float* wp = weight + (o * CIN + i) * NK;
    float s = 0.f;
#pragma unroll
    for (int k = 0; k < NK; ++k) s += pk[k] * wp[k];
    dst[e] = f2bf(s);
  }
}

// ---------------------------------------------------------------------------
// Stage 2: tap-decomposed implicit-GEMM conv via bf16 WMMA.
// One workgroup = 16x32 pixels x 32 out channels for one batch image.
// 8 waves; each wave owns 4 n-tiles (16 pixels each) x 2 m-tiles.
// ---------------------------------------------------------------------------
__global__ __launch_bounds__(256)
void disco_conv_wmma(const float* __restrict__ x,
                     const v16bf* __restrict__ aws,
                     const float* __restrict__ bias,
                     float* __restrict__ out) {
  __shared__ unsigned short lds[LROWS * LCOLS * CHP]; // 66,880 B

  const int tid = threadIdx.x;
  const int bat = blockIdx.z;
  const int h0  = blockIdx.y * TILE_H;
  const int w0  = blockIdx.x * TILE_W;

  // Warm L2 with the (tiny, shared) A-tile workspace.
  __builtin_prefetch((const char*)aws + tid * 384, 0, 1);

  // ---- stage halo tile into LDS as bf16, layout [row][col][ch] -------------
  const int total = LROWS * LCOLS * CIN; // 26,752
  for (int idx = tid; idx < total; idx += 256) {
    const int ch  = idx / (LROWS * LCOLS);
    const int pos = idx - ch * (LROWS * LCOLS);
    const int row = pos / LCOLS;
    const int col = pos - row * LCOLS;
    const int h = h0 - 3 + row;
    const int w = w0 - 3 + col;
    float val = 0.f;
    if ((unsigned)h < (unsigned)H_IMG && (unsigned)w < (unsigned)W_IMG)
      val = x[(((long)bat * CIN + ch) * H_IMG + h) * W_IMG + w];
    lds[(row * LCOLS + col) * CHP + ch] = f2bf(val);
  }
  __syncthreads();

  const int wv   = tid >> 5;     // wave 0..7
  const int lane = tid & 31;
  const int n    = lane & 15;    // B column (pixel within n-tile)
  const int kg   = lane >> 4;    // K half: 0 -> ch 0..7,16..23; 1 -> ch 8..15,24..31

  v8f acc[4][2] = {};

  for (int u = 0; u < KS; ++u) {
    for (int v = 0; v < KS; ++v) {
      const int tap = u * KS + v;
      // A tiles: 32B per lane, pre-swizzled, L2 hot; reused by 8 WMMAs
      v16bf a0 = aws[(tap * 2 + 0) * 32 + lane];
      v16bf a1 = aws[(tap * 2 + 1) * 32 + lane];

#pragma unroll
      for (int t = 0; t < 4; ++t) {
        const int nt = wv * 4 + t;          // n-tile id 0..31
        const int r  = nt >> 1;             // tile row 0..15
        const int cb = (nt & 1) << 4;       // col base 0 or 16
        const int addr = ((r + u) * LCOLS + cb + n + v) * CHP + kg * 8;
        union { uint4 q[2]; v16bf bv; } ub;
        ub.q[0] = *reinterpret_cast<const uint4*>(&lds[addr]);
        ub.q[1] = *reinterpret_cast<const uint4*>(&lds[addr + 16]);
        acc[t][0] = __builtin_amdgcn_wmma_f32_16x16x32_bf16(
            false, a0, false, ub.bv, (short)0, acc[t][0], false, false);
        acc[t][1] = __builtin_amdgcn_wmma_f32_16x16x32_bf16(
            false, a1, false, ub.bv, (short)0, acc[t][1], false, false);
      }
    }
  }

  // ---- epilogue: C/D layout -> out[b][o][h][w], add bias, streaming stores --
#pragma unroll
  for (int t = 0; t < 4; ++t) {
    const int nt = wv * 4 + t;
    const int h  = h0 + (nt >> 1);
    const int w  = w0 + ((nt & 1) << 4) + n;
#pragma unroll
    for (int e = 0; e < 8; ++e) {
      const int o0 = kg * 8 + e;        // M-tile 0
      const int o1 = 16 + kg * 8 + e;   // M-tile 1
      float r0 = acc[t][0][e] + bias[o0];
      float r1 = acc[t][1][e] + bias[o1];
      __builtin_nontemporal_store(
          r0, &out[(((long)bat * COUT + o0) * H_IMG + h) * W_IMG + w]);
      __builtin_nontemporal_store(
          r1, &out[(((long)bat * COUT + o1) * H_IMG + h) * W_IMG + w]);
    }
  }
}

// ---------------------------------------------------------------------------
extern "C" void kernel_launch(void* const* d_in, const int* in_sizes, int n_in,
                              void* d_out, int out_size, void* d_ws, size_t ws_size,
                              hipStream_t stream) {
  const float* x      = (const float*)d_in[0]; // (4,32,1024,1024)
  const float* weight = (const float*)d_in[1]; // (32,32,9)
  const float* bias   = (const float*)d_in[2]; // (32,)
  const float* psi    = (const float*)d_in[3]; // (9,7,7)
  float* out          = (float*)d_out;         // (4,32,1024,1024)

  unsigned short* aws = (unsigned short*)d_ws; // 49*2*32*16 bf16 = 100,352 B

  const int s1_threads = TAPS * 2 * 32;        // 3136
  disco_build_atiles<<<(s1_threads + 255) / 256, 256, 0, stream>>>(weight, psi, aws);

  dim3 grid(W_IMG / TILE_W, H_IMG / TILE_H, 4);
  disco_conv_wmma<<<grid, 256, 0, stream>>>(x, (const v16bf*)d_ws, bias, out);
}
